// FocusedLSTM_88914412962001
// MI455X (gfx1250) — matrix operand order repaired and checked by reading, not compiled, not run on hardware
//
#include <hip/hip_runtime.h>

typedef __attribute__((ext_vector_type(16))) __bf16 v16bf;
typedef __attribute__((ext_vector_type(8)))  __bf16 v8bf;
typedef __attribute__((ext_vector_type(8)))  float  v8f;

#define DI __device__ __forceinline__

constexpr int SEQ = 2048, BATCH = 64, IN = 512, HID = 512;
constexpr int NWG2 = 16;                       // phase-2 persistent workgroups
constexpr int COLS_PER_WG = HID / NWG2;        // 32 columns per WG
constexpr size_t XP_ELEMS = (size_t)SEQ * BATCH * HID;

// ---------------------------------------------------------------------------
// software grid barrier (monotonic counter, one arrive per block per phase)
// ---------------------------------------------------------------------------
DI void grid_barrier(unsigned* cnt, unsigned target) {
  __syncthreads();
  if (threadIdx.x == 0) {
    __threadfence();                                      // release prior stores
    __hip_atomic_fetch_add(cnt, 1u, __ATOMIC_RELAXED, __HIP_MEMORY_SCOPE_AGENT);
    while (__hip_atomic_load(cnt, __ATOMIC_RELAXED, __HIP_MEMORY_SCOPE_AGENT) < target)
      __builtin_amdgcn_s_sleep(1);
  }
  __syncthreads();
  __threadfence();                                        // acquire for all lanes
}

DI float sigmoidf(float x) { return 1.0f / (1.0f + __expf(-x)); }

// ---------------------------------------------------------------------------
// Phase 1: x_proj = x @ W + b   (M=131072, K=512, N=512), bf16 WMMA, f32 acc.
// Writes f32 result into the outputs region of d_out (consumed in-place by
// phase 2, which overwrites it with h). Also resets the grid-barrier counter.
// Block = 256 threads (8 waves), block tile 64(M) x 64(N), K-tile 32.
// ---------------------------------------------------------------------------
__global__ void __launch_bounds__(256) xproj_kernel(
    const float* __restrict__ x, const float* __restrict__ W,
    const float* __restrict__ bias, float* __restrict__ out,
    unsigned* __restrict__ bar)
{
  if (blockIdx.x == 0 && threadIdx.x == 0) *bar = 0u;

  // bf16 fragments staged in LDS in exact WMMA lane layout
  __shared__ __bf16 Alds[4][32][16];   // [m_sub][lane][j]
  __shared__ __bf16 Blds[4][32][16];   // [n_sub][lane][j]

  const int tid  = threadIdx.x;
  const int wave = tid >> 5, lane = tid & 31;
  const int mtile = blockIdx.x >> 3;           // 2048 row tiles of 64
  const int ntile = blockIdx.x & 7;            // 8 col tiles of 64
  const int mbase = mtile * 64;
  const int nbase = ntile * 64;

  const int msub = wave & 3;                   // 4 M-subtiles
  const int ns0  = (wave >> 2) * 2;            // 2 N-subtiles per wave

  v8f acc0 = {}, acc1 = {};

  for (int kt = 0; kt < IN / 32; ++kt) {
#pragma unroll
    for (int i = 0; i < 8; ++i) {
      int e = i * 256 + tid;                   // 0..2047
      {                                        // A: x tile 64 x 32
        int r = e >> 5, kk = e & 31;
        float v = x[(size_t)(mbase + r) * IN + kt * 32 + kk];
        Alds[r >> 4][(((kk >> 3) & 1) << 4) | (r & 15)]
            [(kk & 7) | ((kk >> 4) << 3)] = (__bf16)v;
      }
      {                                        // B: W tile 32 x 64
        int kk = e >> 6, nn = e & 63;
        float v = W[(size_t)(kt * 32 + kk) * HID + nbase + nn];
        Blds[nn >> 4][(((kk >> 3) & 1) << 4) | (nn & 15)]
            [(kk & 7) | ((kk >> 4) << 3)] = (__bf16)v;
      }
    }
    __syncthreads();
    v16bf a  = *(const v16bf*)&Alds[msub][lane][0];
    v16bf b0 = *(const v16bf*)&Blds[ns0][lane][0];
    v16bf b1 = *(const v16bf*)&Blds[ns0 + 1][lane][0];
    acc0 = __builtin_amdgcn_wmma_f32_16x16x32_bf16(false, a, false, b0,
                                                   (short)0, acc0, false, false);
    acc1 = __builtin_amdgcn_wmma_f32_16x16x32_bf16(false, a, false, b1,
                                                   (short)0, acc1, false, false);
    __syncthreads();
  }

  const int kg   = lane >> 4;
  const int col0 = nbase + ns0 * 16 + (lane & 15);
  const float bv0 = bias[col0], bv1 = bias[col0 + 16];
#pragma unroll
  for (int i = 0; i < 8; ++i) {
    size_t row = (size_t)(mbase + msub * 16 + i + 8 * kg);
    out[row * HID + col0]      = acc0[i] + bv0;
    out[row * HID + col0 + 16] = acc1[i] + bv1;
  }
}

// ---------------------------------------------------------------------------
// Phase 2: persistent recurrent scan. 16 WGs x 256 threads; WG g owns 32
// output columns. Ri/Ro column slices live in LDS (bf16, fragment layout) for
// all 2048 steps. h is double-buffered in bf16 in d_ws; one grid barrier per
// step. Each wave owns one 16x16 (batch x col) tile for BOTH gates, so c
// stays in VGPRs the whole time.
// ---------------------------------------------------------------------------
__global__ void __launch_bounds__(256) lstm_scan_kernel(
    const float* __restrict__ h0, const float* __restrict__ c0,
    const float* __restrict__ Ri, const float* __restrict__ Ro,
    const float* __restrict__ bi, const float* __restrict__ bo,
    float* __restrict__ out,            // xp in / h out (in place) + ht/ct tail
    __bf16* __restrict__ hbuf,          // [2][64][512] bf16 in d_ws
    unsigned* __restrict__ bar)
{
  // [gate][kt(16)][n_sub(2)][lane(32)][j(16)]  = 64 KB
  __shared__ __bf16 Blds[2][16][2][32][16];

  const int tid  = threadIdx.x;
  const int wave = tid >> 5, lane = tid & 31;
  const int n0   = blockIdx.x * COLS_PER_WG;

  // ---- preamble: weight slices -> LDS (bf16, WMMA B-fragment layout) ----
  for (int g = 0; g < 2; ++g) {
    const float* R = g ? Ro : Ri;
#pragma unroll
    for (int i = 0; i < 64; ++i) {
      int e = i * 256 + tid;                   // 0..16383
      int k = e >> 5, nn = e & 31;
      float v = R[(size_t)k * HID + n0 + nn];
      Blds[g][k >> 5][nn >> 4][(((k >> 3) & 1) << 4) | (nn & 15)]
          [(k & 7) | (((k >> 4) & 1) << 3)] = (__bf16)v;
    }
  }
  // h0 -> hbuf[0] (own columns only; other WGs fill the rest)
#pragma unroll
  for (int i = 0; i < 8; ++i) {
    int e = i * 256 + tid;                     // 0..2047
    int r = e >> 5, nn = e & 31;
    hbuf[(size_t)r * HID + n0 + nn] = (__bf16)h0[(size_t)r * HID + n0 + nn];
  }

  const int msub = wave & 3;                   // batch subtile (4 x 16 rows)
  const int nsub = wave >> 2;                  // column subtile (2 x 16 cols)
  const int kg   = lane >> 4;
  const int col  = n0 + nsub * 16 + (lane & 15);
  const float biv = bi[col], bov = bo[col];

  float creg[8];
#pragma unroll
  for (int i = 0; i < 8; ++i) {
    int b = msub * 16 + i + 8 * kg;
    creg[i] = c0[(size_t)b * HID + col];
  }

  grid_barrier(bar, NWG2 * 1u);                // preamble visible everywhere

  for (int t = 0; t < SEQ; ++t) {
    const __bf16* hsrc = hbuf + (size_t)(t & 1) * BATCH * HID;
    __bf16*       hdst = hbuf + (size_t)((t + 1) & 1) * BATCH * HID;

    v8f ai = {}, ao = {};
    const __bf16* arow =
        hsrc + (size_t)(msub * 16 + (lane & 15)) * HID + kg * 8;
#pragma unroll 4
    for (int kt = 0; kt < 16; ++kt) {
      v8bf alo = *(const v8bf*)(arow + kt * 32);
      v8bf ahi = *(const v8bf*)(arow + kt * 32 + 16);
      v16bf a  = __builtin_shufflevector(alo, ahi, 0, 1, 2, 3, 4, 5, 6, 7, 8,
                                         9, 10, 11, 12, 13, 14, 15);
      v16bf wbi = *(const v16bf*)&Blds[0][kt][nsub][lane][0];
      v16bf wbo = *(const v16bf*)&Blds[1][kt][nsub][lane][0];
      ai = __builtin_amdgcn_wmma_f32_16x16x32_bf16(false, a, false, wbi,
                                                   (short)0, ai, false, false);
      ao = __builtin_amdgcn_wmma_f32_16x16x32_bf16(false, a, false, wbo,
                                                   (short)0, ao, false, false);
    }

#pragma unroll
    for (int i = 0; i < 8; ++i) {
      int b = msub * 16 + i + 8 * kg;
      size_t idx = ((size_t)t * BATCH + b) * HID + col;
      float xpv = out[idx];                        // x_proj (phase-1 output)
      if (t + 1 < SEQ)                             // warm next step's tile
        __builtin_prefetch(&out[idx + (size_t)BATCH * HID], 0, 1);
      float ig = sigmoidf(ai[i] + biv);
      float og = sigmoidf(ao[i] + bov);
      float cv = creg[i] + ig * tanhf(xpv);
      creg[i] = cv;
      float hv = og * tanhf(cv);
      out[idx] = hv;                               // outputs[t]
      hdst[(size_t)b * HID + col] = (__bf16)hv;    // broadcast for t+1
      if (t == SEQ - 1) {                          // h_t, c_t tails
        out[XP_ELEMS + (size_t)b * HID + col] = hv;
        out[XP_ELEMS + (size_t)BATCH * HID + (size_t)b * HID + col] = cv;
      }
    }
    grid_barrier(bar, (unsigned)(NWG2 * (t + 2)));
  }
}

// ---------------------------------------------------------------------------
extern "C" void kernel_launch(void* const* d_in, const int* in_sizes, int n_in,
                              void* d_out, int out_size, void* d_ws,
                              size_t ws_size, hipStream_t stream) {
  const float* x  = (const float*)d_in[0];
  const float* h0 = (const float*)d_in[1];
  const float* c0 = (const float*)d_in[2];
  const float* W  = (const float*)d_in[3];
  const float* Ri = (const float*)d_in[4];
  const float* Ro = (const float*)d_in[5];
  const float* b  = (const float*)d_in[6];
  const float* bi = (const float*)d_in[7];
  const float* bo = (const float*)d_in[8];
  float* out = (float*)d_out;

  __bf16* hbuf  = (__bf16*)d_ws;                              // 2*64*512*2 B
  unsigned* bar = (unsigned*)((char*)d_ws +
                              2 * BATCH * HID * sizeof(__bf16));

  const int g1 = (SEQ * BATCH / 64) * (HID / 64);             // 16384 blocks
  xproj_kernel<<<g1, 256, 0, stream>>>(x, W, b, out, bar);
  lstm_scan_kernel<<<NWG2, 256, 0, stream>>>(h0, c0, Ri, Ro, bi, bo, out,
                                             hbuf, bar);
}
